// Friday_1760936591622
// MI455X (gfx1250) — matrix operand order repaired and checked by reading, not compiled, not run on hardware
//
#include <hip/hip_runtime.h>
#include <hip/hip_bf16.h>

// ---------------------------------------------------------------------------
// CDNA5 / gfx1250 Hopfield-attention pipeline.
// All matmuls: v_wmma_f32_16x16x32_bf16 (wave32, 16x16 tiles, f32 accum).
// K/V head-slice staging: Tensor Data Mover (tensor_load_to_lds) when the
// builtin exists, with a plain vector-load fallback.
// ---------------------------------------------------------------------------

typedef __attribute__((ext_vector_type(16))) __bf16 v16bf;
typedef __attribute__((ext_vector_type(8)))  float  v8f;

#define HH      8
#define DD      8
#define PP      64        // HH*DD
#define EE      512
#define NSEQ    512
#define BB      32
#define SCALE_F 0.25f
#define LN_EPS  1e-5f

union Frag {
  v16bf v;
  unsigned short u[16];
  uint4 q[2];
};

__device__ __forceinline__ unsigned short f2bf(float f) {
  unsigned int x = __float_as_uint(f);
  x += 0x7fffu + ((x >> 16) & 1u);     // round-to-nearest-even
  return (unsigned short)(x >> 16);
}

__device__ __forceinline__ v8f wmma_bf16(const Frag& a, const Frag& b, v8f c) {
  // (neg_a, A, neg_b, B, c_mod, C, reuse_a, reuse_b)
  return __builtin_amdgcn_wmma_f32_16x16x32_bf16(false, a.v, false, b.v,
                                                 (short)0, c, false, false);
}

// ---------------------------------------------------------------------------
// Tensor Data Mover: 2D strided-gather of a head slice into contiguous LDS.
// D# layout per CDNA5 ISA ch.8 (group0: count/lds_addr/global_addr/type,
// group1: data_size + dims/strides). Issued by one wave; TENSORcnt waited.
// ---------------------------------------------------------------------------
#if __has_builtin(__builtin_amdgcn_tensor_load_to_lds)
#define HAVE_TDM 1
typedef __attribute__((ext_vector_type(4))) unsigned int u32x4;
typedef __attribute__((ext_vector_type(4))) int          i32x4;
typedef __attribute__((ext_vector_type(8))) int          i32x8;

__device__ __forceinline__ void tdm_load_2d(unsigned lds_addr, const void* gaddr,
                                            unsigned tile_d0_dw, unsigned rows,
                                            unsigned row_stride_dw) {
  unsigned long long ga = (unsigned long long)(uintptr_t)gaddr;
  u32x4 g0;
  g0[0] = 1u;                                              // count = 1 (valid D#)
  g0[1] = lds_addr;                                        // LDS byte address
  g0[2] = (unsigned)(ga & 0xffffffffu);                    // global_addr[31:0]
  g0[3] = (unsigned)((ga >> 32) & 0x1ffffffu) | (2u << 30);// addr[56:32] | type=2
  i32x8 g1;
  g1[0] = (int)(2u << 16);                                 // data_size = 4 bytes
  g1[1] = (int)((tile_d0_dw & 0xffffu) << 16);             // tensor_dim0[15:0]
  g1[2] = (int)((rows & 0xffffu) << 16);                   // tensor_dim1[15:0]
  g1[3] = (int)((tile_d0_dw & 0xffffu) << 16);             // tile_dim0
  g1[4] = (int)(rows & 0xffffu);                           // tile_dim1 (tile_dim2=0)
  g1[5] = (int)row_stride_dw;                              // tensor_dim0_stride lo
  g1[6] = 0;
  g1[7] = 0;
  i32x4 z4 = {0, 0, 0, 0};
#if defined(__clang_major__) && (__clang_major__ >= 23)
  i32x8 z8 = {0, 0, 0, 0, 0, 0, 0, 0};
  __builtin_amdgcn_tensor_load_to_lds(g0, g1, z4, z4, z8, 0);
#else
  __builtin_amdgcn_tensor_load_to_lds(g0, g1, z4, z4, 0);
#endif
}

__device__ __forceinline__ void tdm_wait() {
#if __has_builtin(__builtin_amdgcn_s_wait_tensorcnt)
  __builtin_amdgcn_s_wait_tensorcnt(0);
#else
  asm volatile("s_wait_tensorcnt 0x0" ::: "memory");
#endif
}
#else
#define HAVE_TDM 0
#endif

// ---------------------------------------------------------------------------
// Kernel 0: fp32 -> bf16 weight conversion (row-major kept; B-fragments are
// then contiguous 16-element slices of a weight row).
// ---------------------------------------------------------------------------
__global__ void k_cvt_bf16(const float* __restrict__ src,
                           unsigned short* __restrict__ dst, int n) {
  int i = blockIdx.x * blockDim.x + threadIdx.x;
  if (i < n) dst[i] = f2bf(src[i]);
}

// ---------------------------------------------------------------------------
// Kernel 1: fused LayerNorm + Q/K/V projection.
// Block = 16 rows of x (16x512 f32 in LDS), 128 threads = 4 waves.
// Each wave owns one 16-column tile of P=64; K-loop 512/32 = 16 WMMAs.
// One inlined helper per variant so every pointer is a direct kernel arg
// (lets infer-address-spaces emit global_load_* instead of flat_*).
// ---------------------------------------------------------------------------
__device__ __forceinline__ void qkv_one(
    const float (*xs)[EE], unsigned short (*lnbf)[EE],
    const float* mean_s, const float* rstd_s,
    const float* __restrict__ g, const float* __restrict__ b,
    const unsigned short* __restrict__ W, const float* __restrict__ bias,
    unsigned short* __restrict__ out,
    int tid, int lane, int wave, int m0)
{
  const int n = lane & 15, khalf = lane >> 4;
  const int p0 = wave * 16;

  for (int i = tid; i < 16 * EE; i += 128) {
    int r = i >> 9, e = i & 511;
    lnbf[r][e] = f2bf((xs[r][e] - mean_s[r]) * rstd_s[r] * g[e] + b[e]);
  }
  __syncthreads();

  v8f acc = {};
  const unsigned short* wrow = W + (size_t)(p0 + n) * EE;
#pragma unroll 4
  for (int c = 0; c < 16; ++c) {
    Frag a, w;
    int abase = c * 32 + (khalf ? 16 : 0);
    a.q[0] = *(const uint4*)&lnbf[n][abase];          // K 0..7 / 16..23
    a.q[1] = *(const uint4*)&lnbf[n][abase + 8];      // K 8..15 / 24..31
    __builtin_prefetch(wrow + (c + 1) * 32, 0, 1);
    w.v = *(const v16bf*)(wrow + c * 32 + khalf * 16);
    acc = wmma_bf16(a, w, acc);
  }
  float bcol = bias[p0 + n];
#pragma unroll
  for (int r = 0; r < 8; ++r) {
    int row = r + 8 * khalf;
    out[(size_t)(m0 + row) * PP + p0 + n] = f2bf(acc[r] + bcol);
  }
  __syncthreads();
}

__global__ __launch_bounds__(128) void k_qkv(
    const float* __restrict__ x,
    const float* __restrict__ g_q, const float* __restrict__ b_q,
    const float* __restrict__ g_k, const float* __restrict__ b_k,
    const float* __restrict__ g_v, const float* __restrict__ b_v,
    const unsigned short* __restrict__ Wq, const float* __restrict__ bq,
    const unsigned short* __restrict__ Wk, const float* __restrict__ bk,
    const unsigned short* __restrict__ Wv, const float* __restrict__ bv,
    unsigned short* __restrict__ qo, unsigned short* __restrict__ ko,
    unsigned short* __restrict__ vo)
{
  __shared__ alignas(32) float          xs[16][EE];          // 32 KB
  __shared__ alignas(32) unsigned short lnbf[16][EE];        // 16 KB
  __shared__ float red[16][8];
  __shared__ float red2[16][8];
  __shared__ float mean_s[16], rstd_s[16];

  const int tid  = threadIdx.x;
  const int lane = tid & 31;
  const int wave = tid >> 5;
  const int m0   = blockIdx.x * 16;

  // load x tile (16x512 f32) with float4
  {
    const float4* xg  = (const float4*)(x + (size_t)m0 * EE);
    float4*       xsv = (float4*)&xs[0][0];
    for (int i = tid; i < 16 * EE / 4; i += 128) xsv[i] = xg[i];
  }
  __syncthreads();

  // per-row mean / rstd : 8 threads per row, 64 elems each
  {
    int r = tid >> 3, seg = tid & 7;
    float s = 0.f, ss = 0.f;
    for (int e = seg * 64; e < seg * 64 + 64; ++e) {
      float v_ = xs[r][e]; s += v_; ss += v_ * v_;
    }
    red[r][seg] = s; red2[r][seg] = ss;
  }
  __syncthreads();
  if (tid < 16) {
    float s = 0.f, ss = 0.f;
    for (int i = 0; i < 8; ++i) { s += red[tid][i]; ss += red2[tid][i]; }
    float m  = s * (1.0f / EE);
    float vv = ss * (1.0f / EE) - m * m;
    mean_s[tid] = m;
    rstd_s[tid] = rsqrtf(vv + LN_EPS);
  }
  __syncthreads();

  qkv_one(xs, lnbf, mean_s, rstd_s, g_q, b_q, Wq, bq, qo, tid, lane, wave, m0);
  qkv_one(xs, lnbf, mean_s, rstd_s, g_k, b_k, Wk, bk, ko, tid, lane, wave, m0);
  qkv_one(xs, lnbf, mean_s, rstd_s, g_v, b_v, Wv, bv, vo, tid, lane, wave, m0);
}

// ---------------------------------------------------------------------------
// Kernel 2: iterative Hopfield attention, one (batch, head, 16-row tile) per
// block. K/V/probs live in LDS; 3 update steps + final association with V.
// scores use WMMA with K-dim zero-padded (D=8 of 32); a@K / a@V use full
// K=512 split across the 4 waves with a ds_add_f32 reduction.
// ---------------------------------------------------------------------------
__global__ __launch_bounds__(128) void k_attn(
    const unsigned short* __restrict__ qb,
    const unsigned short* __restrict__ kb,
    const unsigned short* __restrict__ vb,
    unsigned short* __restrict__ ob)
{
  __shared__ alignas(32) unsigned short Kbf [NSEQ][8];   // [j][d]   8 KB
  __shared__ alignas(32) unsigned short KbfT[8][NSEQ];   // [d][j]   8 KB
  __shared__ alignas(32) unsigned short VbfT[8][NSEQ];   //          8 KB
  __shared__ alignas(32) unsigned short probs[16][NSEQ]; //         16 KB
  __shared__ alignas(16) unsigned short xi[16][8];
  __shared__ float xacc[16][16];                         //          1 KB
  __shared__ float red[4][16];
  __shared__ float rowstat[16];
#if HAVE_TDM
  __shared__ alignas(32) unsigned short Vstage[NSEQ][8]; //          8 KB
#endif

  const int tid = threadIdx.x, lane = tid & 31, wave = tid >> 5;
  const int rt = blockIdx.x, h = blockIdx.y, bidx = blockIdx.z;
  const size_t base = (size_t)bidx * NSEQ * PP + (size_t)h * DD; // + j*PP + d

#if HAVE_TDM
  // TDM: gather the (NSEQ x 8 bf16) head slices (row stride 128 B) into
  // contiguous LDS.  Wave 0 issues both descriptors; TENSORcnt waited.
  if (wave == 0) {
    tdm_load_2d((unsigned)(uintptr_t)&Kbf[0][0],    kb + base, 4, NSEQ, PP / 2);
    tdm_load_2d((unsigned)(uintptr_t)&Vstage[0][0], vb + base, 4, NSEQ, PP / 2);
    tdm_wait();
  }
  __syncthreads();
  for (int j = tid; j < NSEQ; j += 128) {
    uint4 kv = *(const uint4*)&Kbf[j][0];
    uint4 vv = *(const uint4*)&Vstage[j][0];
    const unsigned short* ksp = (const unsigned short*)&kv;
    const unsigned short* vsp = (const unsigned short*)&vv;
#pragma unroll
    for (int d = 0; d < 8; ++d) { KbfT[d][j] = ksp[d]; VbfT[d][j] = vsp[d]; }
  }
#else
  for (int j = tid; j < NSEQ; j += 128) {
    uint4 kv = *(const uint4*)(kb + base + (size_t)j * PP);
    uint4 vv = *(const uint4*)(vb + base + (size_t)j * PP);
    *(uint4*)&Kbf[j][0] = kv;
    const unsigned short* ksp = (const unsigned short*)&kv;
    const unsigned short* vsp = (const unsigned short*)&vv;
#pragma unroll
    for (int d = 0; d < 8; ++d) { KbfT[d][j] = ksp[d]; VbfT[d][j] = vsp[d]; }
  }
#endif
  if (tid < 16)
    *(uint4*)&xi[tid][0] = *(const uint4*)(qb + base + (size_t)(rt * 16 + tid) * PP);
  __syncthreads();

  const int n = lane & 15, khalf = lane >> 4;

  for (int it = 0; it < 4; ++it) {
    // ---- scores: this wave covers columns j in [wave*128, wave*128+128) ----
    v8f s[8];
    {
      Frag a;
      a.q[0] = make_uint4(0u, 0u, 0u, 0u);
      a.q[1] = make_uint4(0u, 0u, 0u, 0u);
      if (khalf == 0) a.q[0] = *(const uint4*)&xi[n][0];   // K 0..7 valid
#pragma unroll
      for (int t = 0; t < 8; ++t) {
        int j0 = wave * 128 + t * 16;
        Frag kf;
        kf.q[0] = make_uint4(0u, 0u, 0u, 0u);
        kf.q[1] = make_uint4(0u, 0u, 0u, 0u);
        if (khalf == 0) kf.q[0] = *(const uint4*)&Kbf[j0 + n][0];
        v8f z = {};
        s[t] = wmma_bf16(a, kf, z);
      }
    }

    // ---- softmax over j (row-wise): max ----
    float rmax[8];
    {
      float vmax[8];
#pragma unroll
      for (int r = 0; r < 8; ++r) {
        float m = -3.0e38f;
#pragma unroll
        for (int t = 0; t < 8; ++t) m = fmaxf(m, s[t][r]);
        for (int off = 1; off < 16; off <<= 1)
          m = fmaxf(m, __shfl_xor(m, off, 32));
        vmax[r] = m;
      }
      if (lane == 0)  for (int r = 0; r < 8; ++r) red[wave][r]     = vmax[r];
      if (lane == 16) for (int r = 0; r < 8; ++r) red[wave][8 + r] = vmax[r];
      __syncthreads();
      if (tid < 16) {
        float m = red[0][tid];
        for (int w = 1; w < 4; ++w) m = fmaxf(m, red[w][tid]);
        rowstat[tid] = m;
      }
      __syncthreads();
#pragma unroll
      for (int r = 0; r < 8; ++r) rmax[r] = rowstat[r + 8 * khalf];
      __syncthreads();
    }

    // ---- exp + sum ----
    {
      float vsum[8];
#pragma unroll
      for (int r = 0; r < 8; ++r) {
        float ssum = 0.f;
#pragma unroll
        for (int t = 0; t < 8; ++t) {
          float p = __expf((s[t][r] - rmax[r]) * SCALE_F);
          s[t][r] = p;
          ssum += p;
        }
        for (int off = 1; off < 16; off <<= 1)
          ssum += __shfl_xor(ssum, off, 32);
        vsum[r] = ssum;
      }
      if (lane == 0)  for (int r = 0; r < 8; ++r) red[wave][r]     = vsum[r];
      if (lane == 16) for (int r = 0; r < 8; ++r) red[wave][8 + r] = vsum[r];
      __syncthreads();
      if (tid < 16) {
        float ssum = 0.f;
        for (int w = 0; w < 4; ++w) ssum += red[w][tid];
        rowstat[tid] = 1.f / ssum;
      }
      __syncthreads();
    }

    // ---- normalize -> probs (bf16), zero the cross-wave accumulator ----
#pragma unroll
    for (int r = 0; r < 8; ++r) {
      int row = r + 8 * khalf;
      float inv = rowstat[row];
#pragma unroll
      for (int t = 0; t < 8; ++t)
        probs[row][wave * 128 + t * 16 + n] = f2bf(s[t][r] * inv);
    }
    for (int i = tid; i < 256; i += 128) ((float*)xacc)[i] = 0.f;
    __syncthreads();

    // ---- xi_new = probs @ K  (or o = probs @ V on final pass) ----
    // Each wave covers 4 of the 16 K-chunks; reduce via ds_add_f32.
    {
      const unsigned short (*T)[NSEQ] = (it < 3) ? KbfT : VbfT;
      v8f acc = {};
#pragma unroll
      for (int cc = 0; cc < 4; ++cc) {
        int c = wave * 4 + cc;
        Frag a2, b2;
        int abase = c * 32 + (khalf ? 16 : 0);
        a2.q[0] = *(const uint4*)&probs[n][abase];
        a2.q[1] = *(const uint4*)&probs[n][abase + 8];
        b2.q[0] = make_uint4(0u, 0u, 0u, 0u);
        b2.q[1] = make_uint4(0u, 0u, 0u, 0u);
        if (n < 8) b2.v = *(const v16bf*)&T[n][c * 32 + khalf * 16];
        acc = wmma_bf16(a2, b2, acc);
      }
      if (n < 8) {
#pragma unroll
        for (int r = 0; r < 8; ++r)
          atomicAdd(&xacc[r + 8 * khalf][n], acc[r]);   // ds_add_f32
      }
    }
    __syncthreads();

    // ---- commit xi / o from the accumulator (tid -> row,d mapping) ----
    {
      int row = tid >> 3, d = tid & 7;
      if (it < 3) {
        xi[row][d] = f2bf(xacc[row][d]);
      } else {
        ob[base + (size_t)(rt * 16 + row) * PP + d] = f2bf(xacc[row][d]);
      }
    }
    __syncthreads();
  }
}

// ---------------------------------------------------------------------------
// Kernel 3: out-projection (o @ Wo^T + bo) fused with the classifier
// contraction against Wl. Each block = 16 rows (one batch), emits one
// scalar partial; the 32 MB `out` tensor is never materialized.
// ---------------------------------------------------------------------------
__global__ __launch_bounds__(128) void k_oproj(
    const unsigned short* __restrict__ ob,
    const unsigned short* __restrict__ Wo,   // bf16 (E x P) row-major
    const float* __restrict__ bo,
    const float* __restrict__ Wl,            // (1, N*E)
    float* __restrict__ partials)
{
  __shared__ float red[4];
  const int tid = threadIdx.x, lane = tid & 31, wave = tid >> 5;
  const int m0 = blockIdx.x * 16;
  const int n = lane & 15, khalf = lane >> 4;
  float part = 0.f;

  for (int t = wave; t < 32; t += 4) {
    int e0 = t * 16;
    v8f acc = {};
    const unsigned short* wrow = Wo + (size_t)(e0 + n) * PP;
#pragma unroll
    for (int c = 0; c < 2; ++c) {
      Frag a, w;
      int abase = c * 32 + (khalf ? 16 : 0);
      const unsigned short* orow = ob + (size_t)(m0 + n) * PP;
      a.q[0] = *(const uint4*)(orow + abase);
      a.q[1] = *(const uint4*)(orow + abase + 8);
      w.v = *(const v16bf*)(wrow + c * 32 + khalf * 16);
      acc = wmma_bf16(a, w, acc);
    }
    float boe = bo[e0 + n];
#pragma unroll
    for (int r = 0; r < 8; ++r) {
      int row = r + 8 * khalf;
      int m = m0 + row;
      part += (acc[r] + boe) * Wl[(size_t)(m & (NSEQ - 1)) * EE + e0 + n];
    }
  }
  for (int off = 1; off < 32; off <<= 1) part += __shfl_xor(part, off, 32);
  if (lane == 0) red[wave] = part;
  __syncthreads();
  if (tid == 0) partials[blockIdx.x] = red[0] + red[1] + red[2] + red[3];
}

// ---------------------------------------------------------------------------
// Kernel 4: final reduction -> y[b]
// ---------------------------------------------------------------------------
__global__ void k_final(const float* __restrict__ partials,
                        const float* __restrict__ bl,
                        float* __restrict__ y)
{
  int b = threadIdx.x;
  if (b < BB) {
    float s = bl[0];
    for (int i = 0; i < 32; ++i) s += partials[b * 32 + i];
    y[b] = s;
  }
}

// ---------------------------------------------------------------------------
extern "C" void kernel_launch(void* const* d_in, const int* in_sizes, int n_in,
                              void* d_out, int out_size, void* d_ws, size_t ws_size,
                              hipStream_t stream)
{
  const float* x   = (const float*)d_in[0];
  const float* g_q = (const float*)d_in[1];  const float* b_q = (const float*)d_in[2];
  const float* g_k = (const float*)d_in[3];  const float* b_k = (const float*)d_in[4];
  const float* g_v = (const float*)d_in[5];  const float* b_v = (const float*)d_in[6];
  const float* Wq  = (const float*)d_in[7];  const float* bq  = (const float*)d_in[8];
  const float* Wk  = (const float*)d_in[9];  const float* bk  = (const float*)d_in[10];
  const float* Wv  = (const float*)d_in[11]; const float* bv  = (const float*)d_in[12];
  const float* Wo  = (const float*)d_in[13]; const float* bo  = (const float*)d_in[14];
  const float* Wl  = (const float*)d_in[15]; const float* bl  = (const float*)d_in[16];

  char* ws = (char*)d_ws;
  unsigned short* WqB = (unsigned short*)(ws + 0);
  unsigned short* WkB = (unsigned short*)(ws + 65536);
  unsigned short* WvB = (unsigned short*)(ws + 131072);
  unsigned short* WoB = (unsigned short*)(ws + 196608);
  unsigned short* qB  = (unsigned short*)(ws + 262144);
  unsigned short* kB  = (unsigned short*)(ws + 262144 + 1 * 2097152);
  unsigned short* vB  = (unsigned short*)(ws + 262144 + 2 * 2097152);
  unsigned short* oB  = (unsigned short*)(ws + 262144 + 3 * 2097152);
  float* partials     = (float*)(ws + 262144 + 4 * 2097152);

  k_cvt_bf16<<<128, 256, 0, stream>>>(Wq, WqB, PP * EE);
  k_cvt_bf16<<<128, 256, 0, stream>>>(Wk, WkB, PP * EE);
  k_cvt_bf16<<<128, 256, 0, stream>>>(Wv, WvB, PP * EE);
  k_cvt_bf16<<<128, 256, 0, stream>>>(Wo, WoB, EE * PP);

  k_qkv<<<BB * NSEQ / 16, 128, 0, stream>>>(x, g_q, b_q, g_k, b_k, g_v, b_v,
                                            WqB, bq, WkB, bk, WvB, bv,
                                            qB, kB, vB);
  k_attn<<<dim3(NSEQ / 16, HH, BB), 128, 0, stream>>>(qB, kB, vB, oB);
  k_oproj<<<BB * NSEQ / 16, 128, 0, stream>>>(oB, WoB, bo, Wl, partials);
  k_final<<<1, 32, 0, stream>>>(partials, bl, (float*)d_out);
}
